// TransformerLayer_82214263980387
// MI455X (gfx1250) — matrix operand order repaired and checked by reading, not compiled, not run on hardware
//
#include <hip/hip_runtime.h>
#include <hip/hip_bf16.h>

typedef __attribute__((ext_vector_type(16))) _Float16 v16h;
typedef __attribute__((ext_vector_type(8)))  _Float16 v8h;
typedef __attribute__((ext_vector_type(8)))  float    v8f;
typedef __attribute__((ext_vector_type(4)))  unsigned int v4u;
typedef __attribute__((ext_vector_type(8)))  int      v8i;
typedef __attribute__((ext_vector_type(4)))  int      v4i;

// ---------------- constants ----------------
constexpr int Dm  = 1024;
constexpr int Dff = 4096;
constexpr int Sq  = 2048;
constexpr int Bb  = 2;
constexpr int Mrow = Bb * Sq;   // 4096
constexpr int Hn  = 16;

// ---------------- helpers ----------------
__device__ __forceinline__ v16h ld16_split(const _Float16* lo, const _Float16* hi) {
  v8h a = *(const v8h*)lo;
  v8h b = *(const v8h*)hi;
  v16h r;
#pragma unroll
  for (int i = 0; i < 8; ++i) { r[i] = a[i]; r[i + 8] = b[i]; }
  return r;
}
__device__ __forceinline__ v16h ld16_contig(const _Float16* p) {
  return ld16_split(p, p + 8);
}
__device__ __forceinline__ v8f wmma_f16(v16h a, v16h b, v8f c) {
  return __builtin_amdgcn_wmma_f32_16x16x32_f16(false, a, false, b, (short)0, c,
                                                false, false);
}

// ---- Tensor Data Mover: 2-D tile (rows x tile_k halves) global -> LDS ------
// D# layout per CDNA5 ISA.
#if __has_builtin(__builtin_amdgcn_tensor_load_to_lds)
#define HAVE_TDM 1
__device__ __forceinline__ void tdm_load_2d(unsigned int ldsByteOff,
                                            const _Float16* gsrc,
                                            unsigned int tensorD0,   // row length (elems)
                                            unsigned int tensorD1,   // #rows total
                                            unsigned int tileD0,     // tile row len (elems)
                                            unsigned int tileD1,     // tile #rows
                                            unsigned long long stride0) { // row pitch (elems)
  unsigned long long ga = (unsigned long long)(uintptr_t)gsrc;
  v4u g0;
  g0[0] = 1u;                                       // count=1, user mode, no gather
  g0[1] = ldsByteOff;                               // lds_addr (bytes)
  g0[2] = (unsigned int)(ga & 0xffffffffu);         // global_addr[31:0]
  g0[3] = (unsigned int)((ga >> 32) & 0x01ffffffu)  // global_addr[56:32]
          | (2u << 30);                             // type = 2 (image)
  v8i g1;
  g1[0] = (int)(1u << 16);                                  // data_size=1 (2B), wg_mask=0
  g1[1] = (int)((tensorD0 & 0xffffu) << 16);                // tensor_dim0[15:0]
  g1[2] = (int)((tensorD0 >> 16) | ((tensorD1 & 0xffffu) << 16)); // d0 hi | d1 lo
  g1[3] = (int)((tensorD1 >> 16) | (tileD0 << 16));         // d1 hi | tile_dim0
  g1[4] = (int)(tileD1 & 0xffffu);                          // tile_dim1 | tile_dim2=0
  g1[5] = (int)(unsigned int)(stride0 & 0xffffffffu);       // dim0_stride[31:0]
  g1[6] = (int)(unsigned int)((stride0 >> 32) & 0xffffu);   // dim0_stride[47:32]
  g1[7] = 0;
  v4i z = {0, 0, 0, 0};
#if __clang_major__ >= 23
  v8i z8 = {0, 0, 0, 0, 0, 0, 0, 0};
  __builtin_amdgcn_tensor_load_to_lds(g0, g1, z, z, z8, 0);
#else
  __builtin_amdgcn_tensor_load_to_lds(g0, g1, z, z, 0);
#endif
}
#else
#define HAVE_TDM 0
#endif

// ---------------- weight convert + transpose: W[k][n] f32 -> Wt[n][k] f16 ----
__global__ void tr_f32_to_f16T(const float* __restrict__ W, _Float16* __restrict__ Wt,
                               int Kd, int Nd) {
  size_t idx = (size_t)blockIdx.x * 256 + threadIdx.x;
  if (idx >= (size_t)Kd * Nd) return;
  int k = (int)(idx / Nd);
  int n = (int)(idx % Nd);
  Wt[(size_t)n * Kd + k] = (_Float16)W[idx];
}

// ---------------- LayerNorm (ddof=1, eps inside sqrt) -> f16 ----------------
__global__ __launch_bounds__(256) void ln_kernel(const float* __restrict__ x,
                                                 const float* __restrict__ g,
                                                 const float* __restrict__ bt,
                                                 _Float16* __restrict__ out, int Dd) {
  __shared__ float s1[256], s2[256];
  __shared__ float smean, srstd;
  int row = blockIdx.x, tid = threadIdx.x;
  const float* xr = x + (size_t)row * Dd;
  float a = 0.f, b = 0.f;
  for (int i = tid; i < Dd; i += 256) { float v = xr[i]; a += v; b += v * v; }
  s1[tid] = a; s2[tid] = b;
  __syncthreads();
  for (int st = 128; st > 0; st >>= 1) {
    if (tid < st) { s1[tid] += s1[tid + st]; s2[tid] += s2[tid + st]; }
    __syncthreads();
  }
  if (tid == 0) {
    float mean = s1[0] / Dd;
    float var  = (s2[0] - Dd * mean * mean) / (Dd - 1);
    smean = mean;
    srstd = rsqrtf(var + 1e-9f);
  }
  __syncthreads();
  float mean = smean, rstd = srstd;
  for (int i = tid; i < Dd; i += 256)
    out[(size_t)row * Dd + i] = (_Float16)(((xr[i] - mean) * rstd) * g[i] + bt[i]);
}

// ---------------- WMMA GEMM, 128x128 block tile, TDM double-buffered --------
// out = A[M,K] * Bt[N,K]^T + bias (+res)(+silu)
// LDS (dynamic): buf{0,1}: A tile 128x32 f16 (8KB) + B tile 128x32 f16 (8KB)
__global__ __launch_bounds__(256) void gemm_wmma(
    const _Float16* __restrict__ A, const _Float16* __restrict__ Bt,
    const float* __restrict__ bias, const float* __restrict__ res,
    float* __restrict__ outF, _Float16* __restrict__ outH,
    int Mdim, int Ndim, int Kdim, int doSilu) {
  extern __shared__ _Float16 smem[];
  constexpr unsigned BUF_BYTES = 32768u / 2;   // 16KB per buffer
  constexpr int A_HALVES = 128 * 32;           // 4096
  int tid  = threadIdx.x;
  int lane = tid & 31, wv = tid >> 5;
  int wm = wv >> 2, wn = wv & 3;               // 2 x 4 waves, each 64(M) x 32(N)
  int hi = lane >> 4, nl = lane & 15;
  int mBase = blockIdx.y * 128;
  int nBase = blockIdx.x * 128;

  v8f acc[4][2] = {};
  int nk = Kdim >> 5;

#if HAVE_TDM
  if (wv == 0) {
    tdm_load_2d(0u,         A  + (size_t)mBase * Kdim, (unsigned)Kdim,
                (unsigned)Mdim, 32u, 128u, (unsigned long long)Kdim);
    tdm_load_2d(8192u,      Bt + (size_t)nBase * Kdim, (unsigned)Kdim,
                (unsigned)Ndim, 32u, 128u, (unsigned long long)Kdim);
  }
#endif
  for (int i = 0; i < nk; ++i) {
    const _Float16* As = smem + (size_t)(i & 1) * BUF_BYTES;   // halves: (i&1)*16384
    const _Float16* Bs = As + A_HALVES;
#if HAVE_TDM
    if (wv == 0) {
      if (i + 1 < nk) {
        unsigned nb = (unsigned)((i + 1) & 1) * 16384u;
        tdm_load_2d(nb,         A  + (size_t)mBase * Kdim + (i + 1) * 32,
                    (unsigned)Kdim, (unsigned)Mdim, 32u, 128u,
                    (unsigned long long)Kdim);
        tdm_load_2d(nb + 8192u, Bt + (size_t)nBase * Kdim + (i + 1) * 32,
                    (unsigned)Kdim, (unsigned)Ndim, 32u, 128u,
                    (unsigned long long)Kdim);
        __builtin_amdgcn_s_wait_tensorcnt(2);   // current tile pair complete
      } else {
        __builtin_amdgcn_s_wait_tensorcnt(0);
      }
    }
    __syncthreads();
#else
    // fallback: cooperative copy, single-buffer semantics into current buffer
    {
      int row = tid >> 1, cb = (tid & 1) * 16;
      *(v8h*)((_Float16*)As + row * 32 + cb) =
          *(const v8h*)&A[(size_t)(mBase + row) * Kdim + i * 32 + cb];
      *(v8h*)((_Float16*)As + row * 32 + cb + 8) =
          *(const v8h*)&A[(size_t)(mBase + row) * Kdim + i * 32 + cb + 8];
      *(v8h*)((_Float16*)Bs + row * 32 + cb) =
          *(const v8h*)&Bt[(size_t)(nBase + row) * Kdim + i * 32 + cb];
      *(v8h*)((_Float16*)Bs + row * 32 + cb + 8) =
          *(const v8h*)&Bt[(size_t)(nBase + row) * Kdim + i * 32 + cb + 8];
    }
    __syncthreads();
#endif
    v16h af[4];
    v16h bf[2];
#pragma unroll
    for (int t = 0; t < 4; ++t) {
      const _Float16* p = &As[(wm * 64 + t * 16 + nl) * 32 + hi * 8];
      af[t] = ld16_split(p, p + 16);
    }
#pragma unroll
    for (int g = 0; g < 2; ++g)
      bf[g] = ld16_split(&Bs[(wn * 32 + g * 16 + nl) * 32 + hi * 8], &Bs[(wn * 32 + g * 16 + nl) * 32 + hi * 8 + 16]);
#pragma unroll
    for (int t = 0; t < 4; ++t)
#pragma unroll
      for (int g = 0; g < 2; ++g)
        acc[t][g] = wmma_f16(af[t], bf[g], acc[t][g]);
    __syncthreads();
  }
  // epilogue
#pragma unroll
  for (int t = 0; t < 4; ++t)
#pragma unroll
    for (int g = 0; g < 2; ++g) {
      int col = nBase + wn * 32 + g * 16 + nl;
      float bv = bias ? bias[col] : 0.f;
#pragma unroll
      for (int r = 0; r < 8; ++r) {
        int row = mBase + wm * 64 + t * 16 + hi * 8 + r;
        float v = acc[t][g][r] + bv;
        if (doSilu) v = v / (1.f + __expf(-v));
        size_t o = (size_t)row * Ndim + col;
        if (res)  v += res[o];
        if (outF) outF[o] = v;
        if (outH) outH[o] = (_Float16)v;
      }
    }
}

// ---------------- causal flash attention, 1 wave per 16-query tile ----------
__global__ __launch_bounds__(32) void attn_kernel(const _Float16* __restrict__ Q,
                                                  const _Float16* __restrict__ Kh,
                                                  const _Float16* __restrict__ V,
                                                  _Float16* __restrict__ O) {
  __shared__ _Float16 Pt[16 * 32];
  __shared__ _Float16 Vt[64 * 40];
  int lane = threadIdx.x;
  int hi = lane >> 4, nl = lane & 15;
  int qt = blockIdx.x, h = blockIdx.y, b = blockIdx.z;
  size_t headOff = (size_t)b * Sq * Dm + (size_t)h * 64;
  const _Float16* Qb = Q + headOff;
  const _Float16* Kb = Kh + headOff;
  const _Float16* Vb = V + headOff;
  _Float16*       Ob = O + headOff;
  int qBase = qt * 16;

  const _Float16* qp = Qb + (size_t)(qBase + nl) * Dm + hi * 8;
  v16h qf0 = ld16_split(qp,      qp + 16);
  v16h qf1 = ld16_split(qp + 32, qp + 48);

  float mr[8], lr[8];
  v8f acc[4] = {};
#pragma unroll
  for (int r = 0; r < 8; ++r) { mr[r] = -1e30f; lr[r] = 0.f; }

  int nkt = (qBase + 47) >> 5;
  for (int kt = 0; kt < nkt; ++kt) {
    int kt0 = kt << 5;
    v8f c0 = {}, c1 = {};
    const _Float16* kp0 = Kb + (size_t)(kt0 + nl) * Dm + hi * 8;
    c0 = wmma_f16(qf0, ld16_split(kp0, kp0 + 16),      c0);
    c0 = wmma_f16(qf1, ld16_split(kp0 + 32, kp0 + 48), c0);
    const _Float16* kp1 = Kb + (size_t)(kt0 + 16 + nl) * Dm + hi * 8;
    c1 = wmma_f16(qf0, ld16_split(kp1, kp1 + 16),      c1);
    c1 = wmma_f16(qf1, ld16_split(kp1 + 32, kp1 + 48), c1);

    int key0 = kt0 + nl, key1 = kt0 + 16 + nl;
    __syncthreads();
#pragma unroll
    for (int r = 0; r < 8; ++r) {
      int q = qBase + hi * 8 + r;
      float s0 = (key0 <= q) ? c0[r] * 0.125f : -1e30f;
      float s1 = (key1 <= q) ? c1[r] * 0.125f : -1e30f;
      float tm = fmaxf(s0, s1);
#pragma unroll
      for (int off = 1; off < 16; off <<= 1)
        tm = fmaxf(tm, __shfl_xor(tm, off, 32));
      float mnew  = fmaxf(mr[r], tm);
      float scold = __expf(mr[r] - mnew);
      mr[r] = mnew;
      float p0 = __expf(s0 - mnew), p1 = __expf(s1 - mnew);
      float rs = p0 + p1;
#pragma unroll
      for (int off = 1; off < 16; off <<= 1)
        rs += __shfl_xor(rs, off, 32);
      lr[r] = lr[r] * scold + rs;
      acc[0][r] *= scold; acc[1][r] *= scold;
      acc[2][r] *= scold; acc[3][r] *= scold;
      Pt[(hi * 8 + r) * 32 + nl]      = (_Float16)p0;
      Pt[(hi * 8 + r) * 32 + 16 + nl] = (_Float16)p1;
    }
    const _Float16* vp = Vb + (size_t)(kt0 + lane) * Dm;
#pragma unroll
    for (int dc = 0; dc < 8; ++dc) {
      v8h vv = *(const v8h*)(vp + dc * 8);
#pragma unroll
      for (int e = 0; e < 8; ++e) Vt[(dc * 8 + e) * 40 + lane] = vv[e];
    }
    __syncthreads();
    v16h pf = ld16_split(&Pt[nl * 32 + hi * 8], &Pt[nl * 32 + hi * 8 + 16]);
#pragma unroll
    for (int dg = 0; dg < 4; ++dg) {
      v16h vf = ld16_split(&Vt[(dg * 16 + nl) * 40 + hi * 8], &Vt[(dg * 16 + nl) * 40 + hi * 8 + 16]);
      acc[dg] = wmma_f16(pf, vf, acc[dg]);
    }
  }
#pragma unroll
  for (int dg = 0; dg < 4; ++dg)
#pragma unroll
    for (int r = 0; r < 8; ++r) {
      int row = qBase + hi * 8 + r;
      Ob[(size_t)row * Dm + dg * 16 + nl] = (_Float16)(acc[dg][r] / lr[r]);
    }
}

// ---------------- launch ----------------
extern "C" void kernel_launch(void* const* d_in, const int* in_sizes, int n_in,
                              void* d_out, int out_size, void* d_ws, size_t ws_size,
                              hipStream_t stream) {
  const float* x    = (const float*)d_in[0];
  const float* Wq   = (const float*)d_in[1];
  const float* bq   = (const float*)d_in[2];
  const float* Wk   = (const float*)d_in[3];
  const float* bk   = (const float*)d_in[4];
  const float* Wv   = (const float*)d_in[5];
  const float* bv   = (const float*)d_in[6];
  const float* Wo   = (const float*)d_in[7];
  const float* bo   = (const float*)d_in[8];
  const float* ln1g = (const float*)d_in[9];
  const float* ln1b = (const float*)d_in[10];
  const float* ln2g = (const float*)d_in[11];
  const float* ln2b = (const float*)d_in[12];
  const float* Wup  = (const float*)d_in[13];
  const float* bup  = (const float*)d_in[14];
  const float* Wdn  = (const float*)d_in[15];
  const float* bdn  = (const float*)d_in[16];
  float* out = (float*)d_out;

  char* ws = (char*)d_ws;
  auto take = [&](size_t bytes) { char* p = ws; ws += bytes; return p; };
  _Float16* Wqt  = (_Float16*)take((size_t)Dm * Dm * 2);
  _Float16* Wkt  = (_Float16*)take((size_t)Dm * Dm * 2);
  _Float16* Wvt  = (_Float16*)take((size_t)Dm * Dm * 2);
  _Float16* Wot  = (_Float16*)take((size_t)Dm * Dm * 2);
  _Float16* Wupt = (_Float16*)take((size_t)Dm * Dff * 2);
  _Float16* Wdnt = (_Float16*)take((size_t)Dff * Dm * 2);
  _Float16* xn1  = (_Float16*)take((size_t)Mrow * Dm * 2);
  _Float16* Qh   = (_Float16*)take((size_t)Mrow * Dm * 2);
  _Float16* Kh   = (_Float16*)take((size_t)Mrow * Dm * 2);
  _Float16* Vh   = (_Float16*)take((size_t)Mrow * Dm * 2);
  _Float16* pre  = (_Float16*)take((size_t)Mrow * Dm * 2);
  float*    hbuf = (float*)take((size_t)Mrow * Dm * 4);
  _Float16* xn2  = (_Float16*)take((size_t)Mrow * Dm * 2);
  _Float16* up16 = (_Float16*)take((size_t)Mrow * Dff * 2);

  tr_f32_to_f16T<<<(Dm * Dm + 255) / 256, 256, 0, stream>>>(Wq, Wqt, Dm, Dm);
  tr_f32_to_f16T<<<(Dm * Dm + 255) / 256, 256, 0, stream>>>(Wk, Wkt, Dm, Dm);
  tr_f32_to_f16T<<<(Dm * Dm + 255) / 256, 256, 0, stream>>>(Wv, Wvt, Dm, Dm);
  tr_f32_to_f16T<<<(Dm * Dm + 255) / 256, 256, 0, stream>>>(Wo, Wot, Dm, Dm);
  tr_f32_to_f16T<<<(Dm * Dff + 255) / 256, 256, 0, stream>>>(Wup, Wupt, Dm, Dff);
  tr_f32_to_f16T<<<(Dff * Dm + 255) / 256, 256, 0, stream>>>(Wdn, Wdnt, Dff, Dm);

  ln_kernel<<<Mrow, 256, 0, stream>>>(x, ln1g, ln1b, xn1, Dm);

  constexpr size_t GEMM_LDS = 32768;   // 2 x (8KB A + 8KB B)
  dim3 gD(Dm / 128, Mrow / 128);
  gemm_wmma<<<gD, 256, GEMM_LDS, stream>>>(xn1, Wqt, bq, nullptr, nullptr, Qh,
                                           Mrow, Dm, Dm, 0);
  gemm_wmma<<<gD, 256, GEMM_LDS, stream>>>(xn1, Wkt, bk, nullptr, nullptr, Kh,
                                           Mrow, Dm, Dm, 0);
  gemm_wmma<<<gD, 256, GEMM_LDS, stream>>>(xn1, Wvt, bv, nullptr, nullptr, Vh,
                                           Mrow, Dm, Dm, 0);

  attn_kernel<<<dim3(Sq / 16, Hn, Bb), 32, 0, stream>>>(Qh, Kh, Vh, pre);

  gemm_wmma<<<gD, 256, GEMM_LDS, stream>>>(pre, Wot, bo, x, hbuf, nullptr,
                                           Mrow, Dm, Dm, 0);

  ln_kernel<<<Mrow, 256, 0, stream>>>(hbuf, ln2g, ln2b, xn2, Dm);

  gemm_wmma<<<dim3(Dff / 128, Mrow / 128), 256, GEMM_LDS, stream>>>(
      xn2, Wupt, bup, nullptr, nullptr, up16, Mrow, Dff, Dm, 1);

  gemm_wmma<<<gD, 256, GEMM_LDS, stream>>>(up16, Wdnt, bdn, hbuf, out, nullptr,
                                           Mrow, Dm, Dff, 0);
}
